// TemplatePointwiseAttention_10780367913101
// MI455X (gfx1250) — matrix operand order repaired and checked by reading, not compiled, not run on hardware
//
#include <hip/hip_runtime.h>

// ---------------------------------------------------------------------------
// Template pointwise attention, fused, bf16 WMMA (gfx1250 / CDNA5, wave32)
//   C_T=64, C_Z=128, C_HID=16, N_HEADS=4, N_TEMPL=4, N_RES=512
// ---------------------------------------------------------------------------

typedef __attribute__((ext_vector_type(16))) __bf16 v16bf;
typedef __attribute__((ext_vector_type(8)))  __bf16 v8bf;
typedef __attribute__((ext_vector_type(8)))  float  v8f;

#define N_RES   512
#define C_T     64
#define C_Z     128
#define C_HID   16
#define N_HEADS 4
#define N_TEMPL 4
#define NPAIR   (N_RES * N_RES)       // 262144 pairs
#define NTILES  (NPAIR / 16)          // 16384 tiles of 16 pairs
#define WAVES   4
#define BLOCK   (WAVES * 32)
#define GRID    256

__device__ __forceinline__ v8f wmma_bf16(v16bf a, v16bf b, v8f c) {
    // D = A(16x32 bf16) * B(32x16 bf16) + C(16x16 f32)
    return __builtin_amdgcn_wmma_f32_16x16x32_bf16(
        /*neg_a=*/false, a, /*neg_b=*/false, b,
        /*c_mod=*/(short)0, c, /*reuse_a=*/false, /*reuse_b=*/false);
}

// Load an A fragment (16x32 slice, kc-th K chunk) from a per-lane f32 row in
// global memory, converting to bf16. Per ISA A layout:
//   lane l<16 : row l, halves 0..7 = K(kc*32+0..7),  halves 8..15 = K(+16..23)
//   lane l>=16: row l-16, halves 0..7 = K(+8..15),   halves 8..15 = K(+24..31)
__device__ __forceinline__ v16bf load_A_global_f32(const float* __restrict__ row,
                                                   int kc, int lane) {
    const int base = kc * 32 + ((lane >> 4) << 3);
    const float4 a0 = *(const float4*)(row + base + 0);
    const float4 a1 = *(const float4*)(row + base + 4);
    const float4 b0 = *(const float4*)(row + base + 16);
    const float4 b1 = *(const float4*)(row + base + 20);
    v16bf r;
    r[0]  = (__bf16)a0.x; r[1]  = (__bf16)a0.y; r[2]  = (__bf16)a0.z; r[3]  = (__bf16)a0.w;
    r[4]  = (__bf16)a1.x; r[5]  = (__bf16)a1.y; r[6]  = (__bf16)a1.z; r[7]  = (__bf16)a1.w;
    r[8]  = (__bf16)b0.x; r[9]  = (__bf16)b0.y; r[10] = (__bf16)b0.z; r[11] = (__bf16)b0.w;
    r[12] = (__bf16)b1.x; r[13] = (__bf16)b1.y; r[14] = (__bf16)b1.z; r[15] = (__bf16)b1.w;
    return r;
}

// Load an A fragment from a 16x64 row-major bf16 matrix in LDS (the O tile).
__device__ __forceinline__ v16bf load_A_lds(const __bf16* __restrict__ m,
                                            int kc, int lane) {
    const int row  = lane & 15;
    const int base = kc * 32 + ((lane >> 4) << 3);
    union { v16bf v; v8bf h[2]; } u;
    u.h[0] = *(const v8bf*)(m + row * 64 + base);
    u.h[1] = *(const v8bf*)(m + row * 64 + base + 16);
    return u.v;
}

// Load a pre-swizzled B fragment from LDS: frag stored as [lane][16 halves].
__device__ __forceinline__ v16bf load_B(const __bf16* __restrict__ frag, int lane) {
    return *(const v16bf*)(frag + lane * 16);
}

// Store a C-layout accumulator (v8f: lane = column, 8 rows) as bf16 into a
// column-major [cols][16] LDS matrix. One contiguous 16B store per lane.
__device__ __forceinline__ void store_C_colmajor(__bf16* __restrict__ mt, int nc,
                                                 int lane, v8f acc, float scale) {
    const int col     = nc * 16 + (lane & 15);
    const int rowbase = (lane >> 4) << 3;
    v8bf o;
#pragma unroll
    for (int m = 0; m < 8; ++m) o[m] = (__bf16)(acc[m] * scale);
    *(v8bf*)(mt + col * 16 + rowbase) = o;
}

// Stage a row-major f32 weight [KC*32][ncols] into bf16 WMMA B-fragment layout:
// frag(kc,nc) at offset (kc*NC+nc)*512 halves, element [lane*16 + e] where
// lane<16 -> col nc*16+lane, K kc*32+{0..7,16..23}[e] ; lane>=16 -> same col, K kc*32+{8..15,24..31}[e].
__device__ __forceinline__ void stage_w(__bf16* __restrict__ dst,
                                        const float* __restrict__ src,
                                        int KC, int NC, int ncols, int tid) {
    const int total = KC * NC * 512;
    for (int idx = tid; idx < total; idx += BLOCK) {
        const int f  = idx >> 9;
        const int r  = idx & 511;
        const int ln = r >> 4;
        const int e  = r & 15;
        const int kc = f / NC;
        const int nc = f - kc * NC;
        const int col = nc * 16 + (ln & 15);
        const int k   = kc * 32 + ((ln >> 4) << 3) + (e & 7) + ((e >> 3) << 4);
        dst[idx] = (__bf16)src[k * ncols + col];
    }
}

__global__ void __launch_bounds__(BLOCK)
tpa_fused_kernel(const float* __restrict__ t,   // [S, N, N, C_T]
                 const float* __restrict__ z,   // [N, N, C_Z]
                 const float* __restrict__ mask,// [S]
                 const float* __restrict__ wq,  // [C_Z, 64]
                 const float* __restrict__ wk,  // [C_T, 64]
                 const float* __restrict__ wv,  // [C_T, 64]
                 const float* __restrict__ wo,  // [64, C_Z]
                 const float* __restrict__ bo,  // [C_Z]
                 float* __restrict__ out)       // [N, N, C_Z]
{
    // ---- LDS ----------------------------------------------------------------
    __shared__ __attribute__((aligned(32))) __bf16 sWq[16 * 512]; // 4kc x 4nc
    __shared__ __attribute__((aligned(32))) __bf16 sWk[ 8 * 512]; // 2kc x 4nc
    __shared__ __attribute__((aligned(32))) __bf16 sWv[ 8 * 512]; // 2kc x 4nc
    __shared__ __attribute__((aligned(32))) __bf16 sWo[16 * 512]; // 2kc x 8nc
    __shared__ __attribute__((aligned(32))) __bf16 sQT[WAVES][64 * 16];           // col-major
    __shared__ __attribute__((aligned(32))) __bf16 sKT[WAVES][N_TEMPL * 64 * 16]; // col-major/s
    __shared__ __attribute__((aligned(32))) __bf16 sVT[WAVES][N_TEMPL * 64 * 16];
    __shared__ __attribute__((aligned(32))) __bf16 sO [WAVES][16 * 64];           // row-major

    const int tid  = threadIdx.x;
    const int lane = tid & 31;
    const int w    = tid >> 5;

    // ---- one-time weight staging (bf16, B-fragment swizzle) -----------------
    stage_w(sWq, wq, 4, 4,  64, tid);
    stage_w(sWk, wk, 2, 4,  64, tid);
    stage_w(sWv, wv, 2, 4,  64, tid);
    stage_w(sWo, wo, 2, 8, 128, tid);
    __syncthreads();

    const float m0 = mask[0], m1 = mask[1], m2 = mask[2], m3 = mask[3];
    const float bias[N_TEMPL] = {1e9f * (m0 - 1.f), 1e9f * (m1 - 1.f),
                                 1e9f * (m2 - 1.f), 1e9f * (m3 - 1.f)};
    const float gate = (m0 + m1 + m2 + m3 > 0.f) ? 1.f : 0.f;

    // ---- grid-stride loop over 16-pair tiles, one tile per wave -------------
    for (int tile = blockIdx.x * WAVES + w; tile < NTILES; tile += GRID * WAVES) {
        const int pbase = tile * 16;
        const int prow  = pbase + (lane & 15);

        // ---- Q = z * Wq (scaled) : 16 WMMAs ---------------------------------
        const float* zrow = z + (size_t)prow * C_Z;
        v16bf az[4];
#pragma unroll
        for (int kc = 0; kc < 4; ++kc) az[kc] = load_A_global_f32(zrow, kc, lane);
#pragma unroll
        for (int nc = 0; nc < 4; ++nc) {
            v8f acc = {};
#pragma unroll
            for (int kc = 0; kc < 4; ++kc)
                acc = wmma_bf16(az[kc], load_B(sWq + (kc * 4 + nc) * 512, lane), acc);
            store_C_colmajor(sQT[w], nc, lane, acc, 0.25f);  // 1/sqrt(16)
        }

        // ---- K_s, V_s = t_s * Wk / Wv : 64 WMMAs ----------------------------
#pragma unroll
        for (int s = 0; s < N_TEMPL; ++s) {
            const float* trow = t + ((size_t)s * NPAIR + prow) * C_T;
            v16bf at[2];
            at[0] = load_A_global_f32(trow, 0, lane);
            at[1] = load_A_global_f32(trow, 1, lane);
#pragma unroll
            for (int nc = 0; nc < 4; ++nc) {
                v8f ak = {}, av = {};
#pragma unroll
                for (int kc = 0; kc < 2; ++kc) {
                    ak = wmma_bf16(at[kc], load_B(sWk + (kc * 4 + nc) * 512, lane), ak);
                    av = wmma_bf16(at[kc], load_B(sWv + (kc * 4 + nc) * 512, lane), av);
                }
                store_C_colmajor(sKT[w] + s * 1024, nc, lane, ak, 1.f);
                store_C_colmajor(sVT[w] + s * 1024, nc, lane, av, 1.f);
            }
        }
        asm volatile("s_wait_dscnt 0" ::: "memory");

        // ---- softmax over S=4 : 2 (pair,head) items per lane ----------------
#pragma unroll
        for (int it = 0; it < 2; ++it) {
            const int idx = it * 32 + lane;
            const int p   = idx >> 2;
            const int h   = idx & 3;
            float q[16];
#pragma unroll
            for (int d = 0; d < 16; ++d)
                q[d] = (float)sQT[w][(h * 16 + d) * 16 + p];
            float lg[N_TEMPL];
#pragma unroll
            for (int s = 0; s < N_TEMPL; ++s) {
                const __bf16* kcol = sKT[w] + s * 1024;
                float acc = 0.f;
#pragma unroll
                for (int d = 0; d < 16; ++d)
                    acc += q[d] * (float)kcol[(h * 16 + d) * 16 + p];
                lg[s] = acc + bias[s];
            }
            const float mx = fmaxf(fmaxf(lg[0], lg[1]), fmaxf(lg[2], lg[3]));
            float e[N_TEMPL], se = 0.f;
#pragma unroll
            for (int s = 0; s < N_TEMPL; ++s) { e[s] = __expf(lg[s] - mx); se += e[s]; }
            const float inv = 1.f / se;
#pragma unroll
            for (int d = 0; d < 16; ++d) {
                float o = 0.f;
#pragma unroll
                for (int s = 0; s < N_TEMPL; ++s)
                    o += e[s] * (float)sVT[w][s * 1024 + (h * 16 + d) * 16 + p];
                sO[w][p * 64 + h * 16 + d] = (__bf16)(o * inv);
            }
        }
        asm volatile("s_wait_dscnt 0" ::: "memory");

        // ---- out = O * Wo + bo (gated) : 16 WMMAs ---------------------------
        v16bf ao[2];
        ao[0] = load_A_lds(sO[w], 0, lane);
        ao[1] = load_A_lds(sO[w], 1, lane);
#pragma unroll
        for (int nc = 0; nc < 8; ++nc) {
            v8f acc = {};
            acc = wmma_bf16(ao[0], load_B(sWo + (0 * 8 + nc) * 512, lane), acc);
            acc = wmma_bf16(ao[1], load_B(sWo + (1 * 8 + nc) * 512, lane), acc);
            const int col     = nc * 16 + (lane & 15);
            const int rowbase = (lane >> 4) << 3;
            const float b = bo[col];
#pragma unroll
            for (int m = 0; m < 8; ++m)
                out[(size_t)(pbase + rowbase + m) * C_Z + col] = (acc[m] + b) * gate;
        }
    }
}

extern "C" void kernel_launch(void* const* d_in, const int* in_sizes, int n_in,
                              void* d_out, int out_size, void* d_ws, size_t ws_size,
                              hipStream_t stream) {
    const float* t    = (const float*)d_in[0];
    const float* z    = (const float*)d_in[1];
    const float* mask = (const float*)d_in[2];
    const float* wq   = (const float*)d_in[3];
    const float* wk   = (const float*)d_in[4];
    const float* wv   = (const float*)d_in[5];
    const float* wo   = (const float*)d_in[6];
    const float* bo   = (const float*)d_in[7];
    float* out        = (float*)d_out;
    (void)in_sizes; (void)n_in; (void)out_size; (void)d_ws; (void)ws_size;

    tpa_fused_kernel<<<GRID, BLOCK, 0, stream>>>(t, z, mask, wq, wk, wv, wo, bo, out);
}
